// RNNSearch_49632642073063
// MI455X (gfx1250) — matrix operand order repaired and checked by reading, not compiled, not run on hardware
//
#include <hip/hip_runtime.h>
#include <hip/hip_bf16.h>
#include <math.h>

#define B_SZ   32
#define S_LEN  128
#define T_LEN  128
#define H_DIM  1024
#define SCALE_F 32.0f   // sqrt(1024)

typedef __attribute__((ext_vector_type(16))) __bf16 bf16x16;
typedef __attribute__((ext_vector_type(8)))  __bf16 bf16x8;
typedef __attribute__((ext_vector_type(8)))  float  f32x8;

#define WMMA_BF16(a, b, c) \
  __builtin_amdgcn_wmma_f32_16x16x32_bf16(false, (a), false, (b), (short)0, (c), false, false)

__device__ __forceinline__ float sigmoidf_(float x) {
  return 1.0f / (1.0f + __expf(-x));
}

// ---------------------------------------------------------------------------
// One contiguous-K segment of a GEMM: each wave owns TWO adjacent 16-wide
// N-tiles (two accumulators share one A fragment).  Per 32-K tile:
//   A: two aligned 16B loads (bf16x8 lo/hi runs)
//   B: one 32B load per N-tile from fragment-packed weights (w += 512/tile)
//   2 independent v_wmma_f32_16x16x32_bf16
// Fragment layouts used here (contiguous k, same order for A and B):
//   a[j]   <- k = kt*32 + half*16 + j         (j<8)
//   a[8+j] <- k = kt*32 + half*16 + 8 + j
//   b[e]   <- k = kt*32 + half*16 + e,  n = ntile*16 + (lane&15)
//   C/D: row m = half*8 + v, col n = lane&15
// ---------------------------------------------------------------------------
__device__ __forceinline__ void wmma_seg2(f32x8& acc0, f32x8& acc1,
                                          const __bf16* __restrict__ abase, int kTiles,
                                          const __bf16*& w0, const __bf16*& w1,
                                          int half) {
  for (int kt = 0; kt < kTiles; ++kt) {
    bf16x8 lo = *(const bf16x8*)(abase + kt * 32 + half * 16);
    bf16x8 hi = *(const bf16x8*)(abase + kt * 32 + half * 16 + 8);
    bf16x16 a = __builtin_shufflevector(lo, hi, 0, 1, 2, 3, 4, 5, 6, 7,
                                        8, 9, 10, 11, 12, 13, 14, 15);
    bf16x16 b0 = *(const bf16x16*)w0; w0 += 512;
    bf16x16 b1 = *(const bf16x16*)w1; w1 += 512;
    acc0 = WMMA_BF16(a, b0, acc0);
    acc1 = WMMA_BF16(a, b1, acc1);
  }
}

// ------------------------------ utility kernels -----------------------------
// Pack fp32 weight [K,N] row-major into bf16 WMMA-B fragment order:
//   Wp[ ((nt*(K/32)+kt)*32 + lane)*16 + e ] = W[ (kt*32+(lane>>4)*16+e)*N + nt*16+(lane&15) ]
__global__ void pack_w_kernel(const float* __restrict__ W, __bf16* __restrict__ Wp,
                              int K, int N) {
  const size_t total = (size_t)(N / 16) * (K / 32) * 512;
  const int kTiles = K / 32;
  for (size_t idx = (size_t)blockIdx.x * blockDim.x + threadIdx.x; idx < total;
       idx += (size_t)gridDim.x * blockDim.x) {
    const int e    = (int)(idx & 15);
    const int lane = (int)((idx >> 4) & 31);
    const size_t rest = idx >> 9;
    const int kt = (int)(rest % kTiles);
    const int nt = (int)(rest / kTiles);
    const int k = kt * 32 + (lane >> 4) * 16 + e;
    const int n = nt * 16 + (lane & 15);
    Wp[idx] = (__bf16)W[(size_t)k * N + n];
  }
}

__global__ void zero_u32_kernel(unsigned* p, size_t n) {
  size_t i = (size_t)blockIdx.x * blockDim.x + threadIdx.x;
  if (i < n) p[i] = 0u;
}

__global__ void embed_bf16_kernel(const int* __restrict__ toks,
                                  const float* __restrict__ emb,
                                  __bf16* __restrict__ x) {
  const int i   = blockIdx.x;                // token position (b*len + s)
  const int tok = toks[i];
  const float* e = emb + (size_t)tok * H_DIM;
  __bf16* xr = x + (size_t)i * H_DIM;
  for (int h = threadIdx.x; h < H_DIM; h += blockDim.x)
    xr[h] = (__bf16)(e[h] * SCALE_F);
}

__global__ void lens_kernel(const int* __restrict__ enc_in, int* __restrict__ lens) {
  const int b = threadIdx.x;
  if (b < B_SZ) {
    int c = 0;
    for (int s = 0; s < S_LEN; ++s) c += (enc_in[b * S_LEN + s] != 0);
    lens[b] = c;
  }
}

// ------------------------------ encoder GRU ---------------------------------
// gates = sigmoid([x_t , h] @ gk + gb); stores rh = r*h (bf16) and u (f32)
__global__ void enc_gates_kernel(const __bf16* __restrict__ x_all,
                                 const __bf16* __restrict__ h_bf,
                                 const float* __restrict__ h_f,
                                 const __bf16* __restrict__ Wp,
                                 const float* __restrict__ gb,
                                 __bf16* __restrict__ rh_bf,
                                 float* __restrict__ u_f,
                                 const int* __restrict__ lens,
                                 int t, int rev) {
  const int nGroups = (2 * H_DIM) / 32;                    // 64 (2 ntiles each)
  const int wave  = blockIdx.x * 4 + (threadIdx.x >> 5);
  const int lane  = threadIdx.x & 31;
  const int ng    = wave % nGroups;
  const int mtile = wave / nGroups;
  const int mrow  = mtile * 16 + (lane & 15);
  int pos = t;
  if (rev) { const int L = lens[mrow]; pos = (t < L) ? (L - 1 - t) : t; }
  const __bf16* xrow = x_all + ((size_t)mrow * S_LEN + pos) * H_DIM;
  const __bf16* hrow = h_bf + (size_t)mrow * H_DIM;
  const int kT = (2 * H_DIM) / 32;                         // 64
  const __bf16* w0 = Wp + ((size_t)(2 * ng) * kT * 32 + lane) * 16;
  const __bf16* w1 = w0 + (size_t)kT * 512;
  const int half = lane >> 4;
  f32x8 acc[2] = {{}, {}};
  wmma_seg2(acc[0], acc[1], xrow, H_DIM / 32, w0, w1, half);
  wmma_seg2(acc[0], acc[1], hrow, H_DIM / 32, w0, w1, half);
#pragma unroll
  for (int j = 0; j < 2; ++j) {
    const int n = (2 * ng + j) * 16 + (lane & 15);
#pragma unroll
    for (int v = 0; v < 8; ++v) {
      const int row = mtile * 16 + half * 8 + v;
      const float g = sigmoidf_(acc[j][v] + gb[n]);
      if (n < H_DIM)
        rh_bf[(size_t)row * H_DIM + n] = (__bf16)(g * h_f[(size_t)row * H_DIM + n]);
      else
        u_f[(size_t)row * H_DIM + (n - H_DIM)] = g;
    }
  }
}

// candidate + state update + masked/reversed output store
__global__ void enc_cand_kernel(const __bf16* __restrict__ x_all,
                                const __bf16* __restrict__ rh_bf,
                                const float* __restrict__ u_f,
                                const float* __restrict__ h_prev_f,
                                const __bf16* __restrict__ Wp,
                                const float* __restrict__ cb,
                                float* __restrict__ h_next_f,
                                __bf16* __restrict__ h_next_b,
                                float* __restrict__ enc_out_f,
                                __bf16* __restrict__ enc_out_b,
                                const int* __restrict__ lens,
                                const int* __restrict__ enc_in,
                                int t, int rev) {
  const int nGroups = H_DIM / 32;                          // 32
  const int wave  = blockIdx.x * 4 + (threadIdx.x >> 5);
  const int lane  = threadIdx.x & 31;
  const int ng    = wave % nGroups;
  const int mtile = wave / nGroups;
  const int mrow  = mtile * 16 + (lane & 15);
  int pos = t;
  if (rev) { const int L = lens[mrow]; pos = (t < L) ? (L - 1 - t) : t; }
  const __bf16* xrow  = x_all + ((size_t)mrow * S_LEN + pos) * H_DIM;
  const __bf16* rhrow = rh_bf + (size_t)mrow * H_DIM;
  const int kT = (2 * H_DIM) / 32;                         // 64
  const __bf16* w0 = Wp + ((size_t)(2 * ng) * kT * 32 + lane) * 16;
  const __bf16* w1 = w0 + (size_t)kT * 512;
  const int half = lane >> 4;
  f32x8 acc[2] = {{}, {}};
  wmma_seg2(acc[0], acc[1], xrow, H_DIM / 32, w0, w1, half);
  wmma_seg2(acc[0], acc[1], rhrow, H_DIM / 32, w0, w1, half);
#pragma unroll
  for (int j = 0; j < 2; ++j) {
    const int n = (2 * ng + j) * 16 + (lane & 15);
#pragma unroll
    for (int v = 0; v < 8; ++v) {
      const int row = mtile * 16 + half * 8 + v;
      const float c  = tanhf(acc[j][v] + cb[n]);
      const float u  = u_f[(size_t)row * H_DIM + n];
      const float hp = h_prev_f[(size_t)row * H_DIM + n];
      const float hn = u * hp + (1.0f - u) * c;
      const bool  m  = (enc_in[row * S_LEN + t] != 0);
      const float hs = m ? hn : hp;
      h_next_f[(size_t)row * H_DIM + n] = hs;
      h_next_b[(size_t)row * H_DIM + n] = (__bf16)hs;
      int opos = t;
      if (rev) { const int L = lens[row]; opos = (t < L) ? (L - 1 - t) : t; }
      const float ov = m ? hn : 0.0f;
      const size_t oix =
          ((size_t)row * S_LEN + opos) * (2 * H_DIM) + (rev ? H_DIM : 0) + n;
      enc_out_f[oix] = ov;
      enc_out_b[oix] = (__bf16)ov;
    }
  }
}

// ---------------------------- attention pieces ------------------------------
// keys = enc_out[B*S, 2H] @ att_wm[2H, H]
__global__ void keys_kernel(const __bf16* __restrict__ enc_out_b,
                            const __bf16* __restrict__ Wp,
                            float* __restrict__ keys) {
  const int nGroups = H_DIM / 32;                          // 32
  const int wave  = blockIdx.x * 4 + (threadIdx.x >> 5);
  const int lane  = threadIdx.x & 31;
  const int ng    = wave % nGroups;
  const int mtile = wave / nGroups;
  const int mrow  = mtile * 16 + (lane & 15);              // 0..B*S-1
  const __bf16* arow = enc_out_b + (size_t)mrow * 2 * H_DIM;
  const int kT = (2 * H_DIM) / 32;                         // 64
  const __bf16* w0 = Wp + ((size_t)(2 * ng) * kT * 32 + lane) * 16;
  const __bf16* w1 = w0 + (size_t)kT * 512;
  const int half = lane >> 4;
  f32x8 acc[2] = {{}, {}};
  wmma_seg2(acc[0], acc[1], arow, kT, w0, w1, half);
#pragma unroll
  for (int j = 0; j < 2; ++j) {
    const int n = (2 * ng + j) * 16 + (lane & 15);
#pragma unroll
    for (int v = 0; v < 8; ++v) {
      const int row = mtile * 16 + half * 8 + v;
      keys[(size_t)row * H_DIM + n] = acc[j][v];
    }
  }
}

// q = h @ att_wq
__global__ void q_kernel(const __bf16* __restrict__ h_bf,
                         const __bf16* __restrict__ Wp,
                         float* __restrict__ q) {
  const int nGroups = H_DIM / 32;                          // 32
  const int wave  = blockIdx.x * 4 + (threadIdx.x >> 5);
  const int lane  = threadIdx.x & 31;
  const int ng    = wave % nGroups;
  const int mtile = wave / nGroups;
  const int mrow  = mtile * 16 + (lane & 15);
  const __bf16* arow = h_bf + (size_t)mrow * H_DIM;
  const int kT = H_DIM / 32;                               // 32
  const __bf16* w0 = Wp + ((size_t)(2 * ng) * kT * 32 + lane) * 16;
  const __bf16* w1 = w0 + (size_t)kT * 512;
  const int half = lane >> 4;
  f32x8 acc[2] = {{}, {}};
  wmma_seg2(acc[0], acc[1], arow, kT, w0, w1, half);
#pragma unroll
  for (int j = 0; j < 2; ++j) {
    const int n = (2 * ng + j) * 16 + (lane & 15);
#pragma unroll
    for (int v = 0; v < 8; ++v) {
      const int row = mtile * 16 + half * 8 + v;
      q[(size_t)row * H_DIM + n] = acc[j][v];
    }
  }
}

// e = tanh(keys + q) . v ; softmax ; ctx = a @ enc_out   (one block per batch)
__global__ void attn_kernel(const float* __restrict__ keys,
                            const float* __restrict__ q,
                            const float* __restrict__ att_v,
                            const int* __restrict__ enc_in,
                            const float* __restrict__ enc_out_f,
                            __bf16* __restrict__ ctx_b) {
  const int b   = blockIdx.x;
  const int tid = threadIdx.x;                             // blockDim = 128
  __shared__ float e[S_LEN];
  __shared__ float red[S_LEN];
  {
    const int s = tid;
    const float* kr = keys + ((size_t)b * S_LEN + s) * H_DIM;
    const float* qr = q + (size_t)b * H_DIM;
    float sum = 0.0f;
    for (int h = 0; h < H_DIM; ++h) sum += tanhf(kr[h] + qr[h]) * att_v[h];
    e[s] = (enc_in[b * S_LEN + s] != 0) ? sum : -1e30f;
  }
  __syncthreads();
  red[tid] = e[tid];
  __syncthreads();
  for (int off = 64; off > 0; off >>= 1) {
    if (tid < off) red[tid] = fmaxf(red[tid], red[tid + off]);
    __syncthreads();
  }
  const float mx = red[0];
  __syncthreads();
  const float ex = __expf(e[tid] - mx);
  red[tid] = ex;
  __syncthreads();
  for (int off = 64; off > 0; off >>= 1) {
    if (tid < off) red[tid] += red[tid + off];
    __syncthreads();
  }
  const float denom = red[0];
  __syncthreads();
  e[tid] = ex / denom;
  __syncthreads();
  for (int d = tid; d < 2 * H_DIM; d += 128) {
    float acc = 0.0f;
    for (int s = 0; s < S_LEN; ++s)
      acc += e[s] * enc_out_f[((size_t)b * S_LEN + s) * (2 * H_DIM) + d];
    ctx_b[(size_t)b * 2 * H_DIM + d] = (__bf16)acc;
  }
}

// ------------------------------ decoder GRU ---------------------------------
__global__ void dec_gates_kernel(const __bf16* __restrict__ y_all,
                                 const __bf16* __restrict__ ctx_b,
                                 const __bf16* __restrict__ h_bf,
                                 const float* __restrict__ h_f,
                                 const __bf16* __restrict__ Wp,
                                 const float* __restrict__ gb,
                                 __bf16* __restrict__ rh_bf,
                                 float* __restrict__ u_f, int t) {
  const int nGroups = (2 * H_DIM) / 32;                    // 64
  const int wave  = blockIdx.x * 4 + (threadIdx.x >> 5);
  const int lane  = threadIdx.x & 31;
  const int ng    = wave % nGroups;
  const int mtile = wave / nGroups;
  const int mrow  = mtile * 16 + (lane & 15);
  const __bf16* yrow = y_all + ((size_t)mrow * T_LEN + t) * H_DIM;
  const __bf16* crow = ctx_b + (size_t)mrow * 2 * H_DIM;
  const __bf16* hrow = h_bf + (size_t)mrow * H_DIM;
  const int kT = (4 * H_DIM) / 32;                         // 128
  const __bf16* w0 = Wp + ((size_t)(2 * ng) * kT * 32 + lane) * 16;
  const __bf16* w1 = w0 + (size_t)kT * 512;
  const int half = lane >> 4;
  f32x8 acc[2] = {{}, {}};
  wmma_seg2(acc[0], acc[1], yrow, H_DIM / 32, w0, w1, half);
  wmma_seg2(acc[0], acc[1], crow, (2 * H_DIM) / 32, w0, w1, half);
  wmma_seg2(acc[0], acc[1], hrow, H_DIM / 32, w0, w1, half);
#pragma unroll
  for (int j = 0; j < 2; ++j) {
    const int n = (2 * ng + j) * 16 + (lane & 15);
#pragma unroll
    for (int v = 0; v < 8; ++v) {
      const int row = mtile * 16 + half * 8 + v;
      const float g = sigmoidf_(acc[j][v] + gb[n]);
      if (n < H_DIM)
        rh_bf[(size_t)row * H_DIM + n] = (__bf16)(g * h_f[(size_t)row * H_DIM + n]);
      else
        u_f[(size_t)row * H_DIM + (n - H_DIM)] = g;
    }
  }
}

__global__ void dec_cand_kernel(const __bf16* __restrict__ y_all,
                                const __bf16* __restrict__ ctx_b,
                                const __bf16* __restrict__ rh_bf,
                                const float* __restrict__ u_f,
                                const float* __restrict__ h_prev_f,
                                const __bf16* __restrict__ Wp,
                                const float* __restrict__ cb,
                                float* __restrict__ h_next_f,
                                __bf16* __restrict__ h_next_b,
                                float* __restrict__ out, int t) {
  const int nGroups = H_DIM / 32;                          // 32
  const int wave  = blockIdx.x * 4 + (threadIdx.x >> 5);
  const int lane  = threadIdx.x & 31;
  const int ng    = wave % nGroups;
  const int mtile = wave / nGroups;
  const int mrow  = mtile * 16 + (lane & 15);
  const __bf16* yrow  = y_all + ((size_t)mrow * T_LEN + t) * H_DIM;
  const __bf16* crow  = ctx_b + (size_t)mrow * 2 * H_DIM;
  const __bf16* rhrow = rh_bf + (size_t)mrow * H_DIM;
  const int kT = (4 * H_DIM) / 32;                         // 128
  const __bf16* w0 = Wp + ((size_t)(2 * ng) * kT * 32 + lane) * 16;
  const __bf16* w1 = w0 + (size_t)kT * 512;
  const int half = lane >> 4;
  f32x8 acc[2] = {{}, {}};
  wmma_seg2(acc[0], acc[1], yrow, H_DIM / 32, w0, w1, half);
  wmma_seg2(acc[0], acc[1], crow, (2 * H_DIM) / 32, w0, w1, half);
  wmma_seg2(acc[0], acc[1], rhrow, H_DIM / 32, w0, w1, half);
#pragma unroll
  for (int j = 0; j < 2; ++j) {
    const int n = (2 * ng + j) * 16 + (lane & 15);
#pragma unroll
    for (int v = 0; v < 8; ++v) {
      const int row = mtile * 16 + half * 8 + v;
      const float c  = tanhf(acc[j][v] + cb[n]);
      const float u  = u_f[(size_t)row * H_DIM + n];
      const float hp = h_prev_f[(size_t)row * H_DIM + n];
      const float hn = u * hp + (1.0f - u) * c;
      h_next_f[(size_t)row * H_DIM + n] = hn;
      h_next_b[(size_t)row * H_DIM + n] = (__bf16)hn;
      out[((size_t)row * T_LEN + t) * H_DIM + n] = hn;
    }
  }
}

// ------------------------------ host launcher -------------------------------
static inline size_t alignup(size_t x) { return (x + 255) & ~(size_t)255; }

extern "C" void kernel_launch(void* const* d_in, const int* in_sizes, int n_in,
                              void* d_out, int out_size, void* d_ws, size_t ws_size,
                              hipStream_t stream) {
  const int*   enc_in  = (const int*)  d_in[0];
  const int*   dec_in  = (const int*)  d_in[1];
  const float* src_emb = (const float*)d_in[2];
  const float* dst_emb = (const float*)d_in[3];
  const float* fw_gk = (const float*)d_in[4];  const float* fw_gb = (const float*)d_in[5];
  const float* fw_ck = (const float*)d_in[6];  const float* fw_cb = (const float*)d_in[7];
  const float* bw_gk = (const float*)d_in[8];  const float* bw_gb = (const float*)d_in[9];
  const float* bw_ck = (const float*)d_in[10]; const float* bw_cb = (const float*)d_in[11];
  const float* att_wm = (const float*)d_in[12];
  const float* att_wq = (const float*)d_in[13];
  const float* att_v  = (const float*)d_in[14];
  const float* dec_gk = (const float*)d_in[15]; const float* dec_gb = (const float*)d_in[16];
  const float* dec_ck = (const float*)d_in[17]; const float* dec_cb = (const float*)d_in[18];
  float* out = (float*)d_out;

  // ---- workspace layout (bump allocator) ----
  char* ws = (char*)d_ws;
  size_t off = 0;
  auto takeF = [&](size_t n) { float* p = (float*)(ws + off); off = alignup(off + n * 4); return p; };
  auto takeB = [&](size_t n) { __bf16* p = (__bf16*)(ws + off); off = alignup(off + n * 2); return p; };
  auto takeI = [&](size_t n) { int* p = (int*)(ws + off); off = alignup(off + n * 4); return p; };

  int*    lens      = takeI(B_SZ);
  __bf16* x_enc_b   = takeB((size_t)B_SZ * S_LEN * H_DIM);
  __bf16* y_dec_b   = takeB((size_t)B_SZ * T_LEN * H_DIM);
  float*  enc_out_f = takeF((size_t)B_SZ * S_LEN * 2 * H_DIM);
  __bf16* enc_out_b = takeB((size_t)B_SZ * S_LEN * 2 * H_DIM);
  float*  keys_f    = takeF((size_t)B_SZ * S_LEN * H_DIM);
  float*  q_f       = takeF((size_t)B_SZ * H_DIM);
  __bf16* ctx_b     = takeB((size_t)B_SZ * 2 * H_DIM);
  __bf16* rh_b      = takeB((size_t)B_SZ * H_DIM);
  float*  u_f       = takeF((size_t)B_SZ * H_DIM);
  float*  hA_f      = takeF((size_t)B_SZ * H_DIM);
  float*  hB_f      = takeF((size_t)B_SZ * H_DIM);
  __bf16* hA_b      = takeB((size_t)B_SZ * H_DIM);
  __bf16* hB_b      = takeB((size_t)B_SZ * H_DIM);
  __bf16* fw_gk_p  = takeB((size_t)2 * H_DIM * 2 * H_DIM);
  __bf16* fw_ck_p  = takeB((size_t)2 * H_DIM * H_DIM);
  __bf16* bw_gk_p  = takeB((size_t)2 * H_DIM * 2 * H_DIM);
  __bf16* bw_ck_p  = takeB((size_t)2 * H_DIM * H_DIM);
  __bf16* att_wm_p = takeB((size_t)2 * H_DIM * H_DIM);
  __bf16* att_wq_p = takeB((size_t)H_DIM * H_DIM);
  __bf16* dec_gk_p = takeB((size_t)4 * H_DIM * 2 * H_DIM);
  __bf16* dec_ck_p = takeB((size_t)4 * H_DIM * H_DIM);
  if (off > ws_size) return;   // workspace too small; bail deterministically

  auto pack = [&](const float* src, __bf16* dst, int K, int N) {
    const size_t total = (size_t)K * N;
    pack_w_kernel<<<dim3((unsigned)((total + 255) / 256)), 256, 0, stream>>>(src, dst, K, N);
  };
  pack(fw_gk,  fw_gk_p,  2 * H_DIM, 2 * H_DIM);
  pack(fw_ck,  fw_ck_p,  2 * H_DIM, H_DIM);
  pack(bw_gk,  bw_gk_p,  2 * H_DIM, 2 * H_DIM);
  pack(bw_ck,  bw_ck_p,  2 * H_DIM, H_DIM);
  pack(att_wm, att_wm_p, 2 * H_DIM, H_DIM);
  pack(att_wq, att_wq_p, H_DIM, H_DIM);
  pack(dec_gk, dec_gk_p, 4 * H_DIM, 2 * H_DIM);
  pack(dec_ck, dec_ck_p, 4 * H_DIM, H_DIM);

  lens_kernel<<<1, 32, 0, stream>>>(enc_in, lens);
  embed_bf16_kernel<<<B_SZ * S_LEN, 256, 0, stream>>>(enc_in, src_emb, x_enc_b);
  embed_bf16_kernel<<<B_SZ * T_LEN, 256, 0, stream>>>(dec_in, dst_emb, y_dec_b);

  auto zero_h = [&](float* hf, __bf16* hb) {
    const size_t nf = (size_t)B_SZ * H_DIM;            // f32 words
    zero_u32_kernel<<<(unsigned)((nf + 255) / 256), 256, 0, stream>>>((unsigned*)hf, nf);
    const size_t nb = nf / 2;                          // bf16 pairs as u32
    zero_u32_kernel<<<(unsigned)((nb + 255) / 256), 256, 0, stream>>>((unsigned*)hb, nb);
  };

  const dim3 blk(128);                                  // 4 waves / block
  // waves = nGroups * mtiles ; 4 waves per block
  const unsigned gatesBlocks = (64 * 2) / 4;            // 32
  const unsigned candBlocks  = (32 * 2) / 4;            // 16
  const unsigned keysBlocks  = (32 * (B_SZ * S_LEN / 16)) / 4;  // 2048
  const unsigned qBlocks     = (32 * 2) / 4;            // 16

  // ---------------- encoder: forward then backward GRU ----------------
  for (int dir = 0; dir < 2; ++dir) {
    const __bf16* gkp = dir ? bw_gk_p : fw_gk_p;
    const __bf16* ckp = dir ? bw_ck_p : fw_ck_p;
    const float*  gbb = dir ? bw_gb : fw_gb;
    const float*  cbb = dir ? bw_cb : fw_cb;
    zero_h(hA_f, hA_b);
    float* hpf = hA_f; float* hnf = hB_f;
    __bf16* hpb = hA_b; __bf16* hnb = hB_b;
    for (int t = 0; t < S_LEN; ++t) {
      enc_gates_kernel<<<gatesBlocks, blk, 0, stream>>>(x_enc_b, hpb, hpf, gkp, gbb,
                                                        rh_b, u_f, lens, t, dir);
      enc_cand_kernel<<<candBlocks, blk, 0, stream>>>(x_enc_b, rh_b, u_f, hpf, ckp, cbb,
                                                      hnf, hnb, enc_out_f, enc_out_b,
                                                      lens, enc_in, t, dir);
      { float* tf = hpf; hpf = hnf; hnf = tf; }
      { __bf16* tb = hpb; hpb = hnb; hnb = tb; }
    }
  }

  // ---------------- attention keys (one big GEMM) ----------------
  keys_kernel<<<keysBlocks, blk, 0, stream>>>(enc_out_b, att_wm_p, keys_f);

  // ---------------- decoder ----------------
  zero_h(hA_f, hA_b);
  {
    float* hpf = hA_f; float* hnf = hB_f;
    __bf16* hpb = hA_b; __bf16* hnb = hB_b;
    for (int t = 0; t < T_LEN; ++t) {
      q_kernel<<<qBlocks, blk, 0, stream>>>(hpb, att_wq_p, q_f);
      attn_kernel<<<B_SZ, 128, 0, stream>>>(keys_f, q_f, att_v, enc_in, enc_out_f, ctx_b);
      dec_gates_kernel<<<gatesBlocks, blk, 0, stream>>>(y_dec_b, ctx_b, hpb, hpf,
                                                        dec_gk_p, dec_gb, rh_b, u_f, t);
      dec_cand_kernel<<<candBlocks, blk, 0, stream>>>(y_dec_b, ctx_b, rh_b, u_f, hpf,
                                                      dec_ck_p, dec_cb, hnf, hnb, out, t);
      { float* tf = hpf; hpf = hnf; hnf = tf; }
      { __bf16* tb = hpb; hpb = hnb; hnb = tb; }
    }
  }
}
